// Poetry_LM_64742337019996
// MI455X (gfx1250) — compile-verified, matrix operand order checked by reading, not run on hardware
//
#include <hip/hip_runtime.h>
#include <hip/hip_bf16.h>
#include <stdint.h>

#define SEQ   2048
#define EMB   1024
#define HID   1024
#define VOCAB 32000
#define G4    (4*HID)      // 4096 gate width
#define KDIM  1024         // K for both GEMMs (EMB == HID)

// ---- CDNA5 WMMA / TDM types (gfx1250, wave32) ----
typedef __attribute__((ext_vector_type(16))) __bf16 v16bf;
typedef __attribute__((ext_vector_type(8)))  float  v8f;
typedef __attribute__((ext_vector_type(4)))  unsigned int v4u;
typedef __attribute__((ext_vector_type(8)))  int    v8i_;
typedef __attribute__((ext_vector_type(4)))  int    v4i_;

union Frag32B { uint4 q[2]; v16bf v; };   // 32 bytes = 16 bf16 per lane

__device__ __forceinline__ unsigned short f2bf(float f) {
    unsigned int u = __float_as_uint(f);
    unsigned int r = u + 0x7FFFu + ((u >> 16) & 1u);   // round-to-nearest-even
    return (unsigned short)(r >> 16);
}
__device__ __forceinline__ float bf2f(unsigned short s) {
    return __uint_as_float(((unsigned int)s) << 16);
}

// ---------------- prep kernels ----------------
__global__ void k_f32_to_bf16(const float* __restrict__ src,
                              unsigned short* __restrict__ dst, long n) {
    long i = (long)blockIdx.x * blockDim.x + threadIdx.x;
    long stride = (long)gridDim.x * blockDim.x;
    for (; i < n; i += stride) dst[i] = f2bf(src[i]);
}

__global__ void k_bias_sum(const float* __restrict__ a, const float* __restrict__ b,
                           float* __restrict__ out, int n) {
    int i = blockIdx.x * blockDim.x + threadIdx.x;
    if (i < n) out[i] = a[i] + b[i];
}

// one block per token; 256 threads * float4 = 1024 elements
__global__ void k_embed_gather(const int* __restrict__ idx,
                               const float* __restrict__ table,
                               unsigned short* __restrict__ embbf) {
    int t = blockIdx.x;
    int row = idx[t];
    const float4* src = reinterpret_cast<const float4*>(table + (size_t)row * EMB);
    float4 v = src[threadIdx.x];
    unsigned short* d = embbf + (size_t)t * EMB + threadIdx.x * 4;
    d[0] = f2bf(v.x); d[1] = f2bf(v.y); d[2] = f2bf(v.z); d[3] = f2bf(v.w);
}

// ---------------- bf16 WMMA GEMM:  C[M,N] = A[M,K] * B[N,K]^T + bias ----------------
// block = 256 threads = 8 waves; each wave owns TWO 16x16 D tiles (shared A frag, two
// independent WMMA accumulation chains); block covers 16(M) x 256(N).
// The full 16x1024 bf16 A panel (32KB of the 320KB/WGP LDS) is DMA'd once by the
// Tensor Data Mover; the K loop then runs with zero barriers.
__global__ __launch_bounds__(256)
void k_gemm_bf16_wmma(const unsigned short* __restrict__ A,   // M x K row-major bf16
                      const unsigned short* __restrict__ B,   // N x K row-major bf16
                      const float* __restrict__ bias,         // N
                      float* __restrict__ C,                  // M x N f32
                      int M, int N, int do_relu) {
    __shared__ unsigned short Atile[16 * KDIM];   // 32 KB

    const int tid  = threadIdx.x;
    const int lane = tid & 31;
    const int wave = tid >> 5;
    const int mbase  = blockIdx.y * 16;
    const int nbase0 = (blockIdx.x * 16 + wave * 2) * 16;
    const int nbase1 = nbase0 + 16;

    // ---- stage A panel: TDM (tensor_load_to_lds) from wave 0, others wait at barrier
#if __has_builtin(__builtin_amdgcn_tensor_load_to_lds)
    if (tid < 32) {   // one wave issues the DMA (TDM ignores EXEC; avoid 8 redundant DMAs)
        unsigned long long ga = (unsigned long long)(uintptr_t)(A + (size_t)mbase * KDIM);
        unsigned ldsOff = (unsigned)(size_t)(void*)Atile;
        v4u g0;
        g0.x = 1u;                                          // count=1, user descriptor
        g0.y = ldsOff;                                      // lds_addr
        g0.z = (unsigned)(ga & 0xFFFFFFFFu);                // global_addr[31:0]
        g0.w = (unsigned)((ga >> 32) & 0x01FFFFFFu) | (2u << 30);  // global[56:32] | type=2
        v8i_ g1;
        g1[0] = 0x00010000;            // workgroup_mask=0, data_size=1 (2 bytes)
        g1[1] = (int)(KDIM << 16);     // tensor_dim0 = 1024
        g1[2] = (int)(16 << 16);       // tensor_dim1 = 16
        g1[3] = (int)(KDIM << 16);     // tile_dim0 = 1024
        g1[4] = 16;                    // tile_dim1 = 16 (tile_dim2 = 0)
        g1[5] = KDIM;                  // tensor_dim0_stride = 1024
        g1[6] = 0;
        g1[7] = 0;
        v4i_ z4 = {0, 0, 0, 0};
#if __clang_major__ >= 23
        v8i_ z8 = {0, 0, 0, 0, 0, 0, 0, 0};
        __builtin_amdgcn_tensor_load_to_lds(g0, g1, z4, z4, z8, 0);
#else
        __builtin_amdgcn_tensor_load_to_lds(g0, g1, z4, z4, 0);
#endif
        __builtin_amdgcn_s_wait_tensorcnt(0);
    }
#else
    // fallback: cooperative load of the full panel (16x1024 bf16 = 2048 uint4)
    for (int u = tid; u < (16 * KDIM) / 8; u += 256) {
        int arow = u >> 7;            // /128 chunks-per-row
        int acol = (u & 127) * 8;
        *reinterpret_cast<uint4*>(&Atile[arow * KDIM + acol]) =
            *reinterpret_cast<const uint4*>(A + (size_t)(mbase + arow) * KDIM + acol);
    }
#endif
    __syncthreads();

    // 16-bit A/B fragment striping (ISA 7.12.2): lane<16 -> K chunks {0..7,16..23},
    // lane>=16 -> {8..15,24..31}; row/col = lane&15.
    const int rc   = lane & 15;
    const int koff = (lane >> 4) * 8;

    const unsigned short* Brow0 = B + (size_t)(nbase0 + rc) * KDIM;
    const unsigned short* Brow1 = B + (size_t)(nbase1 + rc) * KDIM;

    v8f acc0 = {};
    v8f acc1 = {};
    #pragma unroll 4
    for (int kk = 0; kk < KDIM; kk += 32) {
        if ((kk & 127) == 0 && kk + 128 < KDIM) {      // global_prefetch_b8 ahead
            __builtin_prefetch(Brow0 + kk + 128, 0, 3);
            __builtin_prefetch(Brow1 + kk + 128, 0, 3);
        }
        Frag32B a, b0, b1;
        a.q[0]  = *reinterpret_cast<const uint4*>(&Atile[rc * KDIM + kk + koff]);
        a.q[1]  = *reinterpret_cast<const uint4*>(&Atile[rc * KDIM + kk + koff + 16]);
        b0.q[0] = *reinterpret_cast<const uint4*>(Brow0 + kk + koff);
        b0.q[1] = *reinterpret_cast<const uint4*>(Brow0 + kk + koff + 16);
        b1.q[0] = *reinterpret_cast<const uint4*>(Brow1 + kk + koff);
        b1.q[1] = *reinterpret_cast<const uint4*>(Brow1 + kk + koff + 16);
        acc0 = __builtin_amdgcn_wmma_f32_16x16x32_bf16(
            false, a.v, false, b0.v, (short)0, acc0, false, false);
        acc1 = __builtin_amdgcn_wmma_f32_16x16x32_bf16(
            false, a.v, false, b1.v, (short)0, acc1, false, false);
    }

    // D layout: col = lane&15, rows = (lane>>4)*8 + v
    int col0 = nbase0 + (lane & 15);
    int col1 = nbase1 + (lane & 15);
    int rowb = mbase + (lane >> 4) * 8;
    float bv0 = bias ? bias[col0] : 0.0f;
    float bv1 = bias ? bias[col1] : 0.0f;
    #pragma unroll
    for (int v = 0; v < 8; ++v) {
        float x0 = acc0[v] + bv0;
        float x1 = acc1[v] + bv1;
        if (do_relu) { x0 = fmaxf(x0, 0.0f); x1 = fmaxf(x1, 0.0f); }
        C[(size_t)(rowb + v) * N + col0] = x0;
        C[(size_t)(rowb + v) * N + col1] = x1;
    }
}

// ---------------- LSTM recurrence: persistent single workgroup (32 wave32s) ----------------
__global__ __launch_bounds__(1024)
void k_lstm_scan(const float* __restrict__ xW,           // SEQ x 4096 (x@W_ih.T + b)
                 const unsigned short* __restrict__ Whh, // 4096 x 1024 bf16
                 unsigned short* __restrict__ hsbf,      // SEQ x HID bf16 (feeds logits GEMM)
                 float* __restrict__ out_hc) {           // [hT(1024), cT(1024)]
    __shared__ float hsh[HID];
    const int i = threadIdx.x;
    hsh[i] = 0.0f;
    float c = 0.0f, h = 0.0f;
    __syncthreads();

    const unsigned short* w0 = Whh + (size_t)(0 * HID + i) * HID;  // gate i
    const unsigned short* w1 = Whh + (size_t)(1 * HID + i) * HID;  // gate f
    const unsigned short* w2 = Whh + (size_t)(2 * HID + i) * HID;  // gate g
    const unsigned short* w3 = Whh + (size_t)(3 * HID + i) * HID;  // gate o

    for (int t = 0; t < SEQ; ++t) {
        const float* xwr = xW + (size_t)t * G4;
        float a0 = xwr[i];
        float a1 = xwr[HID + i];
        float a2 = xwr[2 * HID + i];
        float a3 = xwr[3 * HID + i];

        for (int k = 0; k < HID; k += 8) {
            uint4 q0 = *reinterpret_cast<const uint4*>(w0 + k);
            uint4 q1 = *reinterpret_cast<const uint4*>(w1 + k);
            uint4 q2 = *reinterpret_cast<const uint4*>(w2 + k);
            uint4 q3 = *reinterpret_cast<const uint4*>(w3 + k);
            const unsigned short* s0 = reinterpret_cast<const unsigned short*>(&q0);
            const unsigned short* s1 = reinterpret_cast<const unsigned short*>(&q1);
            const unsigned short* s2 = reinterpret_cast<const unsigned short*>(&q2);
            const unsigned short* s3 = reinterpret_cast<const unsigned short*>(&q3);
            #pragma unroll
            for (int j = 0; j < 8; ++j) {
                float hk = hsh[k + j];
                a0 = fmaf(bf2f(s0[j]), hk, a0);
                a1 = fmaf(bf2f(s1[j]), hk, a1);
                a2 = fmaf(bf2f(s2[j]), hk, a2);
                a3 = fmaf(bf2f(s3[j]), hk, a3);
            }
        }

        float ig = 1.0f / (1.0f + __expf(-a0));
        float fg = 1.0f / (1.0f + __expf(-a1));
        float gg = tanhf(a2);
        float og = 1.0f / (1.0f + __expf(-a3));
        c = fg * c + ig * gg;
        h = og * tanhf(c);

        __syncthreads();
        hsh[i] = h;
        hsbf[(size_t)t * HID + i] = f2bf(h);
        __syncthreads();
    }
    out_hc[i]       = h;   // hT
    out_hc[HID + i] = c;   // cT
}

// ---------------- log-softmax: per-row max + log-sum-exp, then subtract ----------------
__global__ __launch_bounds__(256)
void k_row_lse(const float* __restrict__ logits, float* __restrict__ lse) {
    __shared__ float red[256];
    int row = blockIdx.x;
    const float* p = logits + (size_t)row * VOCAB;

    float mx = -3.4e38f;
    for (int c = threadIdx.x; c < VOCAB; c += 256) mx = fmaxf(mx, p[c]);
    red[threadIdx.x] = mx; __syncthreads();
    for (int s = 128; s > 0; s >>= 1) {
        if (threadIdx.x < s) red[threadIdx.x] = fmaxf(red[threadIdx.x], red[threadIdx.x + s]);
        __syncthreads();
    }
    mx = red[0]; __syncthreads();

    float sum = 0.0f;
    for (int c = threadIdx.x; c < VOCAB; c += 256) sum += __expf(p[c] - mx);
    red[threadIdx.x] = sum; __syncthreads();
    for (int s = 128; s > 0; s >>= 1) {
        if (threadIdx.x < s) red[threadIdx.x] += red[threadIdx.x + s];
        __syncthreads();
    }
    if (threadIdx.x == 0) lse[row] = mx + __logf(red[0]);
}

__global__ void k_sub_lse(float* __restrict__ out, const float* __restrict__ lse) {
    int row = blockIdx.y;
    int col = blockIdx.x * blockDim.x + threadIdx.x;
    out[(size_t)row * VOCAB + col] -= lse[row];
}

// ---------------- host launcher ----------------
extern "C" void kernel_launch(void* const* d_in, const int* in_sizes, int n_in,
                              void* d_out, int out_size, void* d_ws, size_t ws_size,
                              hipStream_t stream) {
    const int*   input_x = (const int*)  d_in[0];
    const float* emb     = (const float*)d_in[1];
    const float* W_ih    = (const float*)d_in[2];
    const float* W_hh    = (const float*)d_in[3];
    const float* b_ih    = (const float*)d_in[4];
    const float* b_hh    = (const float*)d_in[5];
    const float* W_lin   = (const float*)d_in[6];
    const float* b_lin   = (const float*)d_in[7];
    float* out = (float*)d_out;

    // workspace carve-out (~124 MB total)
    char* ws = (char*)d_ws;
    size_t off = 0;
    auto carve = [&](size_t bytes) -> void* {
        void* p = (void*)(ws + off);
        off += (bytes + 255) & ~(size_t)255;
        return p;
    };
    unsigned short* Wihb  = (unsigned short*)carve((size_t)G4 * EMB * 2);
    unsigned short* Whhb  = (unsigned short*)carve((size_t)G4 * HID * 2);
    unsigned short* Wlinb = (unsigned short*)carve((size_t)VOCAB * HID * 2);
    unsigned short* embb  = (unsigned short*)carve((size_t)SEQ * EMB * 2);
    unsigned short* hsbf  = (unsigned short*)carve((size_t)SEQ * HID * 2);
    float* xW   = (float*)carve((size_t)SEQ * G4 * 4);
    float* bsum = (float*)carve((size_t)G4 * 4);
    float* lse  = (float*)carve((size_t)SEQ * 4);
    (void)ws_size; (void)in_sizes; (void)n_in; (void)out_size;

    // 1) cast weights to bf16 (L2-resident operand packs), fold biases
    k_f32_to_bf16<<<512, 256, 0, stream>>>(W_ih,  Wihb,  (long)G4 * EMB);
    k_f32_to_bf16<<<512, 256, 0, stream>>>(W_hh,  Whhb,  (long)G4 * HID);
    k_f32_to_bf16<<<2048, 256, 0, stream>>>(W_lin, Wlinb, (long)VOCAB * HID);
    k_bias_sum<<<(G4 + 255) / 256, 256, 0, stream>>>(b_ih, b_hh, bsum, G4);

    // 2) embedding gather -> bf16 activations
    k_embed_gather<<<SEQ, 256, 0, stream>>>(input_x, emb, embb);

    // 3) xW = X @ W_ih^T + (b_ih+b_hh)   [2048 x 4096]
    dim3 g1(G4 / 256, SEQ / 16);
    k_gemm_bf16_wmma<<<g1, 256, 0, stream>>>(embb, Wihb, bsum, xW, SEQ, G4, 0);

    // 4) sequential LSTM scan (writes hT/cT into tail of d_out)
    k_lstm_scan<<<1, 1024, 0, stream>>>(xW, Whhb, hsbf, out + (size_t)SEQ * VOCAB);

    // 5) logits = relu(hs @ W_lin^T + b_lin)   [2048 x 32000]
    dim3 g2(VOCAB / 256, SEQ / 16);
    k_gemm_bf16_wmma<<<g2, 256, 0, stream>>>(hsbf, Wlinb, b_lin, out, SEQ, VOCAB, 1);

    // 6) log-softmax: per-row LSE then subtract
    k_row_lse<<<SEQ, 256, 0, stream>>>(out, lse);
    dim3 g3(VOCAB / 256, SEQ);
    k_sub_lse<<<g3, 256, 0, stream>>>(out, lse);
}
